// DARNN_72361609003102
// MI455X (gfx1250) — matrix operand order repaired and checked by reading, not compiled, not run on hardware
//
#include <hip/hip_runtime.h>

typedef unsigned short u16;
typedef unsigned int   u32;

typedef __attribute__((ext_vector_type(16))) __bf16 v16bf;
typedef __attribute__((ext_vector_type(8)))  float  v8f;

#define BATCH 1024
#define TT    256
#define NF    128
#define HE    256
#define HD    256
#define TP1   257
#define TP1P  288     // 257 padded to multiple of 32 (K-pad)
#define NP384 384     // 257 padded to multiple of 128 (N-pad)

// ---------- scalar helpers ----------
__device__ __forceinline__ u16 f2bf(float f) {
  u32 u = __float_as_uint(f);
  u32 r = u + 0x7FFFu + ((u >> 16) & 1u);   // round-to-nearest-even
  return (u16)(r >> 16);
}
__device__ __forceinline__ float bf2f(u16 h) {
  return __uint_as_float(((u32)h) << 16);
}
__device__ __forceinline__ float sigm(float x) { return 1.0f / (1.0f + __expf(-x)); }
// branch-free tanh: saturates correctly (exp->inf => 1, exp->0 => -1)
__device__ __forceinline__ float fast_tanh(float x) {
  return 1.0f - 2.0f / (__expf(2.0f * x) + 1.0f);
}

// ---------- tiled bf16 WMMA GEMM:  out = act( X[M,K] @ W[N,K]^T + bias ) ----------
// REQUIRES: M % 64 == 0, N % 128 == 0, K % 32 == 0 (caller pads buffers).
// X, W bf16 row-major (ld = K). Output row-major (ld = N), f32 and/or bf16.
#define BM 64
#define BN 128
#define BK 32
#define XS_LD (BK + 8)
#define WS_LD (BK + 8)

__device__ __forceinline__ void stage_tiles(const u16* __restrict__ X,
                                            const u16* __restrict__ W,
                                            int K, int m0, int n0, int k0, int tid,
                                            u16* __restrict__ Xs, u16* __restrict__ Ws) {
  const int r = tid >> 2;            // 0..63
  const int c = (tid & 3) * 8;       // 0,8,16,24
  uint4 xv  = *(const uint4*)(X + (size_t)(m0 + r) * K + (k0 + c));
  uint4 wv0 = *(const uint4*)(W + (size_t)(n0 + r) * K + (k0 + c));
  uint4 wv1 = *(const uint4*)(W + (size_t)(n0 + r + 64) * K + (k0 + c));
  *(uint4*)&Xs[r * XS_LD + c]        = xv;
  *(uint4*)&Ws[r * WS_LD + c]        = wv0;
  *(uint4*)&Ws[(r + 64) * WS_LD + c] = wv1;
}

__global__ __launch_bounds__(256)
void gemm_bf16(const u16* __restrict__ X, const u16* __restrict__ W,
               const float* __restrict__ bias,
               float* __restrict__ outF, u16* __restrict__ outH,
               int M, int N, int K, int act)
{
  __shared__ u16 Xs[2][BM * XS_LD];
  __shared__ u16 Ws[2][BN * WS_LD];

  const int tid  = threadIdx.x;
  const int lane = tid & 31;
  const int wave = tid >> 5;       // 0..7
  const int wm   = wave >> 2;      // 0..1  -> M direction
  const int wn   = wave & 3;       // 0..3  -> N direction
  const int m0   = blockIdx.y * BM;
  const int n0   = blockIdx.x * BN;
  const int half = lane >> 4;      // 0/1
  const int lid  = lane & 15;

  v8f acc[2][2] = {};

  const int nk = K / BK;
  stage_tiles(X, W, K, m0, n0, 0, tid, Xs[0], Ws[0]);

  for (int kt = 0; kt < nk; ++kt) {
    const int cur = kt & 1;
    __syncthreads();
    if (kt + 1 < nk)
      stage_tiles(X, W, K, m0, n0, (kt + 1) * BK, tid, Xs[cur ^ 1], Ws[cur ^ 1]);

    // per-lane fragment gather per ISA 7.12.2 (16-bit A 16x32, B 32x16)
    union Frag { v16bf v; u16 u[16]; };
    Frag A[2], Bf[2];
#pragma unroll
    for (int mi = 0; mi < 2; ++mi) {
      const u16* row = &Xs[cur][(wm * 32 + mi * 16 + lid) * XS_LD];
#pragma unroll
      for (int v = 0; v < 8; ++v) {
        int kk = ((v & 4) ? 16 : 0) + half * 8 + (v & 3) * 2;
        A[mi].u[2 * v]     = row[kk];
        A[mi].u[2 * v + 1] = row[kk + 1];
      }
    }
#pragma unroll
    for (int ni = 0; ni < 2; ++ni) {
      const u16* row = &Ws[cur][(wn * 32 + ni * 16 + lid) * WS_LD];
#pragma unroll
      for (int v = 0; v < 8; ++v) {
        int kk = ((v & 4) ? 16 : 0) + half * 8 + (v & 3) * 2;
        Bf[ni].u[2 * v]     = row[kk];
        Bf[ni].u[2 * v + 1] = row[kk + 1];
      }
    }
#pragma unroll
    for (int mi = 0; mi < 2; ++mi)
#pragma unroll
      for (int ni = 0; ni < 2; ++ni)
        acc[mi][ni] = __builtin_amdgcn_wmma_f32_16x16x32_bf16(
            false, A[mi].v, false, Bf[ni].v, (short)0, acc[mi][ni], false, false);
  }

  // store: D layout -> M = v + 8*half, N = lid (no guards: dims tile-exact)
#pragma unroll
  for (int mi = 0; mi < 2; ++mi)
#pragma unroll
    for (int ni = 0; ni < 2; ++ni) {
      int cm = m0 + wm * 32 + mi * 16 + 8 * half;
      int cn = n0 + wn * 32 + ni * 16 + lid;
      float bv = bias[cn];
#pragma unroll
      for (int v = 0; v < 8; ++v) {
        int gm = cm + v;
        float r = acc[mi][ni][v] + bv;
        if (act) r = fast_tanh(r);
        if (outF) outF[(size_t)gm * N + cn] = r;
        if (outH) outH[(size_t)gm * N + cn] = f2bf(r);
      }
    }
}

// ---------- one-time weight prep (zero-padded bf16 copies) ----------
__global__ void k_pad_w(const float* __restrict__ src, u16* __restrict__ dst,
                        int Rs, int Cs, int Rd, int Cd) {
  int i = blockIdx.x * 256 + threadIdx.x;
  if (i >= Rd * Cd) return;
  int r = i / Cd, c = i % Cd;
  float v = (r < Rs && c < Cs) ? src[r * Cs + c] : 0.0f;
  dst[i] = f2bf(v);
}
__global__ void k_pad_bias(const float* __restrict__ src, float* __restrict__ dst,
                           int ns, int nd) {
  int i = blockIdx.x * 256 + threadIdx.x;
  if (i < nd) dst[i] = (i < ns) ? src[i] : 0.0f;
}
__global__ void k_pack_wge(const float* __restrict__ Wih, const float* __restrict__ Whh,
                           u16* __restrict__ dst) {      // [4HE, NF+HE] = [1024,384]
  int i = blockIdx.x * 256 + threadIdx.x;
  if (i >= 1024 * 384) return;
  int r = i / 384, c = i % 384;
  float v = (c < NF) ? Wih[r * NF + c] : Whh[r * HE + (c - NF)];
  dst[i] = f2bf(v);
}
__global__ void k_pack_wgd(const float* __restrict__ Wih, const float* __restrict__ Whh,
                           u16* __restrict__ dst) {      // [4HD, TP1P] = [1024,288]
  int i = blockIdx.x * 256 + threadIdx.x;
  if (i >= 1024 * TP1P) return;
  int r = i / TP1P, c = i % TP1P;
  float v = (c == 0) ? Wih[r] : (c <= HD ? Whh[r * HD + (c - 1)] : 0.0f);
  dst[i] = f2bf(v);
}
__global__ void k_bias_sum(const float* a, const float* b, float* d, int n) {
  int i = blockIdx.x * 256 + threadIdx.x;
  if (i < n) d[i] = a[i] + b[i];
}
__global__ void k_zero(float* p, int n) {
  int i = blockIdx.x * 256 + threadIdx.x;
  if (i < n) p[i] = 0.0f;
}

// ---------- per-step pack kernels (concat -> contiguous bf16 X) ----------
__global__ void k_pack_enc_attn(const float* __restrict__ feat, const float* __restrict__ h,
                                const float* __restrict__ c, u16* __restrict__ X, int t) {
  int i = blockIdx.x * 256 + threadIdx.x;                // BATCH*640
  if (i >= BATCH * 640) return;
  int b = i / 640, col = i % 640;
  float v;
  if (col < NF)            v = feat[(size_t)b * (TP1 * NF) + (size_t)t * NF + col];
  else if (col < NF + HE)  v = h[b * HE + (col - NF)];
  else                     v = c[b * HE + (col - NF - HE)];
  X[i] = f2bf(v);
}
__global__ void k_pack_enc_gates(const float* __restrict__ a, const float* __restrict__ feat,
                                 const float* __restrict__ h, u16* __restrict__ X, int t) {
  int i = blockIdx.x * 256 + threadIdx.x;                // BATCH*384
  if (i >= BATCH * 384) return;
  int b = i / 384, col = i % 384;
  float v;
  if (col < NF) v = a[b * NF + col] * feat[(size_t)b * (TP1 * NF) + (size_t)t * NF + col];
  else          v = h[b * HE + (col - NF)];
  X[i] = f2bf(v);
}
__global__ void k_pack_dec_attn(const float* __restrict__ h, const float* __restrict__ c,
                                const u16* __restrict__ hs, u16* __restrict__ X, int t) {
  int i = blockIdx.x * 256 + threadIdx.x;                // BATCH*768
  if (i >= BATCH * 768) return;
  int b = i / 768, col = i % 768;
  u16 v;
  if (col < HD)            v = f2bf(h[b * HD + col]);
  else if (col < 2 * HD)   v = f2bf(c[b * HD + (col - HD)]);
  else                     v = hs[(size_t)b * (TP1 * HE) + (size_t)t * HE + (col - 2 * HD)];
  X[i] = v;
}
__global__ void k_pack_dec_gates(const float* __restrict__ xi, const float* __restrict__ h,
                                 u16* __restrict__ X) {  // BATCH x TP1P (288), zero pad
  int i = blockIdx.x * 256 + threadIdx.x;
  if (i >= BATCH * TP1P) return;
  int b = i / TP1P, col = i % TP1P;
  float v = (col == 0) ? xi[b] : (col <= HD ? h[b * HD + (col - 1)] : 0.0f);
  X[i] = f2bf(v);
}

// ---------- LSTM pointwise (gates order i,f,g,o), branch-free transcendentals ----------
__global__ __launch_bounds__(256)
void k_lstm(const float* __restrict__ gates, float* __restrict__ c, float* __restrict__ h,
            u16* __restrict__ hsOut, int t) {
  int idx = blockIdx.x * 256 + threadIdx.x;              // BATCH*256
  int b = idx >> 8, j = idx & 255;
  size_t gb = (size_t)b * 1024;
  float gi = gates[gb + j];
  float gf = gates[gb + 256 + j];
  float gg = gates[gb + 512 + j];
  float go = gates[gb + 768 + j];
  float cn = sigm(gf) * c[idx] + sigm(gi) * fast_tanh(gg);
  float hn = sigm(go) * fast_tanh(cn);
  c[idx] = cn;
  h[idx] = hn;
  if (hsOut) hsOut[(size_t)b * (TP1 * HE) + (size_t)t * HE + j] = f2bf(hn);
}

// ---------- row softmax over 257 valid cols (row stride NP384) ----------
__global__ __launch_bounds__(256)
void k_softmax(float* __restrict__ w) {
  int b = blockIdx.x, tid = threadIdx.x;
  float* row = w + (size_t)b * NP384;
  float v0 = row[tid];
  float v1 = (tid == 0) ? row[256] : -3.4e38f;
  __shared__ float red[256];
  red[tid] = fmaxf(v0, v1);
  __syncthreads();
  for (int st = 128; st > 0; st >>= 1) { if (tid < st) red[tid] = fmaxf(red[tid], red[tid + st]); __syncthreads(); }
  float m = red[0];
  __syncthreads();
  float e0 = __expf(v0 - m);
  float e1 = (tid == 0) ? __expf(v1 - m) : 0.0f;
  red[tid] = e0 + e1;
  __syncthreads();
  for (int st = 128; st > 0; st >>= 1) { if (tid < st) red[tid] += red[tid + st]; __syncthreads(); }
  float inv = 1.0f / red[0];
  row[tid] = e0 * inv;
  if (tid == 0) row[256] = e1 * inv;
}

// ---------- context einsum bt,bth->bh (hs in bf16 / L2) + fused comb_fc ----------
__global__ __launch_bounds__(256)
void k_context(const float* __restrict__ w, const u16* __restrict__ hs,
               const float* __restrict__ target, const float* __restrict__ Wc,
               const float* __restrict__ bc, float* __restrict__ feati,
               float* __restrict__ xi, int t) {
  int b = blockIdx.x, h = threadIdx.x;
  const float* wr = w + (size_t)b * NP384;
  const u16*  hb = hs + (size_t)b * (TP1 * HE) + h;
  float acc = 0.0f;
  for (int s = 0; s < TP1; ++s) {
    if (s + 8 < TP1) __builtin_prefetch(hb + (size_t)(s + 8) * HE, 0, 1);
    acc += wr[s] * bf2f(hb[(size_t)s * HE]);
  }
  feati[(size_t)b * HE + h] = acc;
  __shared__ float red[256];
  red[h] = acc * Wc[1 + h];
  __syncthreads();
  for (int st = 128; st > 0; st >>= 1) { if (h < st) red[h] += red[h + st]; __syncthreads(); }
  if (h == 0) xi[b] = target[(size_t)b * TT + t] * Wc[0] + red[0] + bc[0];
}

// ---------- final projection out[b] = [h_d, feat_last] . Wf + bf ----------
__global__ __launch_bounds__(256)
void k_final(const float* __restrict__ hD, const float* __restrict__ feati,
             const float* __restrict__ Wf, const float* __restrict__ bf,
             float* __restrict__ out) {
  int b = blockIdx.x, j = threadIdx.x;
  float s = hD[b * HD + j] * Wf[j] + feati[b * HE + j] * Wf[HD + j];
  __shared__ float red[256];
  red[j] = s;
  __syncthreads();
  for (int st = 128; st > 0; st >>= 1) { if (j < st) red[j] += red[j + st]; __syncthreads(); }
  if (j == 0) out[b] = red[0] + bf[0];
}

// =======================================================================
extern "C" void kernel_launch(void* const* d_in, const int* in_sizes, int n_in,
                              void* d_out, int out_size, void* d_ws, size_t ws_size,
                              hipStream_t stream) {
  (void)in_sizes; (void)n_in; (void)out_size; (void)ws_size;
  const float* feat     = (const float*)d_in[0];
  const float* target   = (const float*)d_in[1];
  const float* enc_Wih  = (const float*)d_in[2];
  const float* enc_Whh  = (const float*)d_in[3];
  const float* enc_bih  = (const float*)d_in[4];
  const float* enc_bhh  = (const float*)d_in[5];
  const float* enc_Wa1  = (const float*)d_in[6];
  const float* enc_ba1  = (const float*)d_in[7];
  const float* enc_Wa2  = (const float*)d_in[8];
  const float* enc_ba2  = (const float*)d_in[9];
  const float* dec_Wih  = (const float*)d_in[10];
  const float* dec_Whh  = (const float*)d_in[11];
  const float* dec_bih  = (const float*)d_in[12];
  const float* dec_bhh  = (const float*)d_in[13];
  const float* dec_Wa1  = (const float*)d_in[14];
  const float* dec_ba1  = (const float*)d_in[15];
  const float* dec_Wa2  = (const float*)d_in[16];
  const float* dec_ba2  = (const float*)d_in[17];
  const float* dec_Wc   = (const float*)d_in[18];
  const float* dec_bc   = (const float*)d_in[19];
  const float* dec_Wf   = (const float*)d_in[20];
  const float* dec_bf   = (const float*)d_in[21];

  // ---- workspace carve-out (256B aligned) ----
  char* base = (char*)d_ws;
  size_t off = 0;
  auto carve = [&](size_t bytes) -> void* {
    void* p = base + off;
    off += (bytes + 255) & ~(size_t)255;
    return p;
  };
  u16*   wa1e  = (u16*)  carve((size_t)NP384 * 640 * 2);   // [384,640] rows>=257 zero
  u16*   wa2e  = (u16*)  carve((size_t)128 * NP384 * 2);   // [128,384] cols>=257 zero
  u16*   wge   = (u16*)  carve((size_t)1024 * 384 * 2);    // [1024,384]
  u16*   wa1d  = (u16*)  carve((size_t)256 * 768 * 2);     // [256,768]
  u16*   wa2d  = (u16*)  carve((size_t)NP384 * 256 * 2);   // [384,256] rows>=257 zero
  u16*   wgd   = (u16*)  carve((size_t)1024 * TP1P * 2);   // [1024,288] cols>=257 zero
  float* benc  = (float*)carve(1024 * 4);
  float* bdec  = (float*)carve(1024 * 4);
  float* ba1p  = (float*)carve(NP384 * 4);                 // enc_ba1 padded to 384
  float* ba2dp = (float*)carve(NP384 * 4);                 // dec_ba2 padded to 384
  float* hE    = (float*)carve((size_t)BATCH * HE * 4);
  float* cE    = (float*)carve((size_t)BATCH * HE * 4);
  float* hD    = (float*)carve((size_t)BATCH * HD * 4);
  float* cD    = (float*)carve((size_t)BATCH * HD * 4);
  u16*   hsB   = (u16*)  carve((size_t)BATCH * TP1 * HE * 2);  // 134.7 MB, L2-resident
  u16*   Xa    = (u16*)  carve((size_t)BATCH * 768 * 2);
  u16*   A1    = (u16*)  carve((size_t)BATCH * NP384 * 2);
  float* aEnc  = (float*)carve((size_t)BATCH * NF * 4);
  u16*   Xg    = (u16*)  carve((size_t)BATCH * 384 * 2);
  float* gates = (float*)carve((size_t)BATCH * 1024 * 4);
  float* logit = (float*)carve((size_t)BATCH * NP384 * 4);
  float* feati = (float*)carve((size_t)BATCH * HE * 4);
  float* xi    = (float*)carve((size_t)BATCH * 4);

  dim3 blk(256);

  // ---- one-time prep (deterministic each call) ----
  int n;
  n = NP384 * 640; k_pad_w<<<(n + 255) / 256, blk, 0, stream>>>(enc_Wa1, wa1e, 257, 640, NP384, 640);
  n = 128 * NP384; k_pad_w<<<(n + 255) / 256, blk, 0, stream>>>(enc_Wa2, wa2e, 128, 257, 128, NP384);
  n = 256 * 768;   k_pad_w<<<(n + 255) / 256, blk, 0, stream>>>(dec_Wa1, wa1d, 256, 768, 256, 768);
  n = NP384 * 256; k_pad_w<<<(n + 255) / 256, blk, 0, stream>>>(dec_Wa2, wa2d, 257, 256, NP384, 256);
  k_pack_wge<<<(1024 * 384 + 255) / 256, blk, 0, stream>>>(enc_Wih, enc_Whh, wge);
  k_pack_wgd<<<(1024 * TP1P + 255) / 256, blk, 0, stream>>>(dec_Wih, dec_Whh, wgd);
  k_bias_sum<<<4, blk, 0, stream>>>(enc_bih, enc_bhh, benc, 1024);
  k_bias_sum<<<4, blk, 0, stream>>>(dec_bih, dec_bhh, bdec, 1024);
  k_pad_bias<<<2, blk, 0, stream>>>(enc_ba1, ba1p, 257, NP384);
  k_pad_bias<<<2, blk, 0, stream>>>(dec_ba2, ba2dp, 257, NP384);
  n = BATCH * HE;
  k_zero<<<(n + 255) / 256, blk, 0, stream>>>(hE, n);
  k_zero<<<(n + 255) / 256, blk, 0, stream>>>(cE, n);
  k_zero<<<(n + 255) / 256, blk, 0, stream>>>(hD, n);
  k_zero<<<(n + 255) / 256, blk, 0, stream>>>(cD, n);

  // ---- encoder: T+1 steps ----
  for (int t = 0; t < TP1; ++t) {
    k_pack_enc_attn<<<(BATCH * 640) / 256, blk, 0, stream>>>(feat, hE, cE, Xa, t);
    gemm_bf16<<<dim3(3, 16), blk, 0, stream>>>(Xa, wa1e, ba1p, nullptr, A1,
                                               BATCH, NP384, 640, 1);
    gemm_bf16<<<dim3(1, 16), blk, 0, stream>>>(A1, wa2e, enc_ba2, aEnc, nullptr,
                                               BATCH, 128, NP384, 0);
    k_pack_enc_gates<<<(BATCH * 384) / 256, blk, 0, stream>>>(aEnc, feat, hE, Xg, t);
    gemm_bf16<<<dim3(8, 16), blk, 0, stream>>>(Xg, wge, benc, gates, nullptr,
                                               BATCH, 1024, 384, 0);
    k_lstm<<<1024, blk, 0, stream>>>(gates, cE, hE, hsB, t);
  }

  // ---- decoder: T steps ----
  for (int t = 0; t < TT; ++t) {
    k_pack_dec_attn<<<(BATCH * 768) / 256, blk, 0, stream>>>(hD, cD, hsB, Xa, t);
    gemm_bf16<<<dim3(2, 16), blk, 0, stream>>>(Xa, wa1d, dec_ba1, nullptr, A1,
                                               BATCH, 256, 768, 1);
    gemm_bf16<<<dim3(3, 16), blk, 0, stream>>>(A1, wa2d, ba2dp, logit, nullptr,
                                               BATCH, NP384, 256, 0);
    k_softmax<<<1024, blk, 0, stream>>>(logit);
    k_context<<<1024, blk, 0, stream>>>(logit, hsB, target, dec_Wc, dec_bc, feati, xi, t);
    k_pack_dec_gates<<<(BATCH * TP1P) / 256, blk, 0, stream>>>(xi, hD, Xg);
    gemm_bf16<<<dim3(8, 16), blk, 0, stream>>>(Xg, wgd, bdec, gates, nullptr,
                                               BATCH, 1024, TP1P, 0);
    k_lstm<<<1024, blk, 0, stream>>>(gates, cD, hD, nullptr, 0);
  }

  k_final<<<1024, blk, 0, stream>>>(hD, feati, dec_Wf, dec_bf, (float*)d_out);
}
